// CosmosAttentionBlock_70866960384116
// MI455X (gfx1250) — matrix operand order repaired and checked by reading, not compiled, not run on hardware
//
#include <hip/hip_runtime.h>
#include <hip/hip_bf16.h>
#include <cstdint>
#include <cstddef>

// ---------------------------------------------------------------------------
// CDNA5 / gfx1250 transformer block:  LN -> QKV -> attention -> proj(+res)
//                                     -> LN -> FC1(GELU) -> FC2(+res)
// GEMMs: v_wmma_f32_16x16x32_bf16 (wave32 WMMA, fp32 accum),
//        double-buffered LDS fed by global_load_async_to_lds_b128 (ASYNCcnt),
//        one barrier per K-step, global_prefetch_b8 warming L2.
// Block tile 256x128, 8 waves, each wave owns a 64x64 slab = 4x4 WMMA tiles.
// ---------------------------------------------------------------------------

typedef __bf16 bf16;
typedef __attribute__((ext_vector_type(16))) __bf16 v16bf;
typedef __attribute__((ext_vector_type(8)))  float  v8f;

union FragBF {
    uint4 u[2];   // 32 bytes
    v16bf v;      // 16 bf16
};
union PackBF {
    uint4 u;      // 16 bytes
    bf16  h[8];
};

constexpr int BM  = 256;      // block rows
constexpr int BN  = 128;      // block cols
constexpr int BK  = 32;       // K-step (== WMMA K)
constexpr int STR = BK + 8;   // padded LDS row stride (40 halves = 80B, 16B-aligned)
constexpr int MT  = 4;        // 16x16 M-subtiles per wave
constexpr int NT  = 4;        // 16x16 N-subtiles per wave

__device__ __forceinline__ uint32_t lds_addr(const void* p)
{
    // low 32 bits of a generic pointer to __shared__ == LDS byte offset
    return (uint32_t)(uintptr_t)p;
}

__device__ __forceinline__ void async_copy_b128(uint32_t lds_off, const void* gptr)
{
    asm volatile("global_load_async_to_lds_b128 %0, %1, off"
                 :: "v"(lds_off), "v"((uint64_t)(uintptr_t)gptr)
                 : "memory");
}

__device__ __forceinline__ void wait_async0()
{
    asm volatile("s_wait_asynccnt 0x0" ::: "memory");
}

template <bool TRANS_B>
__global__ __launch_bounds__(256)
void gemm_wmma_kernel(const bf16* __restrict__ A,  int lda, long long strideA,
                      const bf16* __restrict__ Bm, int ldb, long long strideB,
                      const float* __restrict__ bias,
                      const float* __restrict__ resid,
                      float* __restrict__ Cf, bf16* __restrict__ Cb,
                      int ldc, long long strideC,
                      int K, float alpha, int doGelu)
{
    __shared__ bf16 sA[2][BM * STR];   // 2 x 20 KB
    __shared__ bf16 sB[2][BN * STR];   // 2 x 10 KB (60 KB total)

    const int tid = threadIdx.x;
    const int m0  = blockIdx.y * BM;
    const int n0  = blockIdx.x * BN;
    const long long z = blockIdx.z;
    A  += z * strideA;
    Bm += z * strideB;
    if (Cf) Cf += z * strideC;
    if (Cb) Cb += z * strideC;

    const int wave  = tid >> 5;
    const int lane  = tid & 31;
    const int waveM = wave >> 1;   // 0..3 : 64-row slab
    const int waveN = wave & 1;    // 0..1 : 64-col slab
    const int laneN = lane & 15;
    const int laneH = lane >> 4;   // K-half / M-half selector

    // ---- async staging of one K-tile into buffer `buf` ----
    auto stageA = [&](int buf, int k0) {
#pragma unroll
        for (int rep = 0; rep < 4; ++rep) {
            int u = tid + rep * 256;
            int r = u >> 2;
            int c = (u & 3) * 8;
            async_copy_b128(lds_addr(&sA[buf][r * STR + c]),
                            &A[(size_t)(m0 + r) * lda + k0 + c]);
        }
    };
    auto stageB = [&](int buf, int k0) {
        if (!TRANS_B) {
            // B = W row-major (N x K): straight async copy
#pragma unroll
            for (int rep = 0; rep < 2; ++rep) {
                int u = tid + rep * 256;
                int r = u >> 2;
                int c = (u & 3) * 8;
                async_copy_b128(lds_addr(&sB[buf][r * STR + c]),
                                &Bm[(size_t)(n0 + r) * ldb + k0 + c]);
            }
        } else {
            // B row-major (K x N): coalesced b128 along N, transpose on LDS write
#pragma unroll
            for (int rep = 0; rep < 2; ++rep) {
                int e  = rep * 256 + tid;      // 0..511
                int kk = e >> 4;               // 0..31
                int n8 = (e & 15) * 8;         // 0..120
                PackBF pk;
                pk.u = *(const uint4*)&Bm[(size_t)(k0 + kk) * ldb + n0 + n8];
#pragma unroll
                for (int j = 0; j < 8; ++j)
                    sB[buf][(n8 + j) * STR + kk] = pk.h[j];
            }
        }
    };

    v8f acc[MT][NT];
#pragma unroll
    for (int mt = 0; mt < MT; ++mt)
#pragma unroll
        for (int nt = 0; nt < NT; ++nt)
#pragma unroll
            for (int i = 0; i < 8; ++i) acc[mt][nt][i] = 0.0f;

    const int nk = K / BK;

    // prologue: tile 0 -> buffer 0
    stageA(0, 0);
    stageB(0, 0);

    for (int kt = 0; kt < nk; ++kt) {
        // own async copies for tile kt complete (in-order), then make visible
        wait_async0();
        __syncthreads();

        if (kt + 1 < nk) {
            int k1 = (kt + 1) * BK;
            // L2 warm-up one more tile ahead
            if (kt + 2 < nk) {
                int k2 = (kt + 2) * BK;
                {
                    int r = tid >> 2, c = (tid & 3) * 8;
                    __builtin_prefetch(&A[(size_t)(m0 + r) * lda + k2 + c], 0, 1);
                }
                if (!TRANS_B) {
                    int r = tid >> 2, c = (tid & 3) * 8;
                    __builtin_prefetch(&Bm[(size_t)(n0 + r) * ldb + k2 + c], 0, 1);
                } else {
                    int kk = tid >> 4, n8 = (tid & 15) * 8;
                    __builtin_prefetch(&Bm[(size_t)(k2 + kk) * ldb + n0 + n8], 0, 1);
                }
            }
            // stage tile kt+1 into the other buffer (its last readers finished
            // compute(kt-1), which every wave completed before this barrier)
            stageA((kt + 1) & 1, k1);
            stageB((kt + 1) & 1, k1);
        }

        const bf16* __restrict__ cA = sA[kt & 1];
        const bf16* __restrict__ cB = sB[kt & 1];

        // ---- fragments per documented CDNA5 WMMA VGPR layouts ----
        FragBF af[MT], bfr[NT];
#pragma unroll
        for (int mt = 0; mt < MT; ++mt) {
            // A: lane holds row (lane&15); K chunks {laneH*8..+7} and {16+laneH*8..+7}
            int row = waveM * 64 + mt * 16 + laneN;
            int o   = row * STR + laneH * 8;
            af[mt].u[0] = *(const uint4*)&cA[o];
            af[mt].u[1] = *(const uint4*)&cA[o + 16];
        }
#pragma unroll
        for (int nt = 0; nt < NT; ++nt) {
            // B: lane holds column (lane&15); K chunks {laneH*8..+7} and {16+laneH*8..+7}
            int col = waveN * 64 + nt * 16 + laneN;
            int o   = col * STR + laneH * 8;
            bfr[nt].u[0] = *(const uint4*)&cB[o];
            bfr[nt].u[1] = *(const uint4*)&cB[o + 16];
        }

#pragma unroll
        for (int mt = 0; mt < MT; ++mt)
#pragma unroll
            for (int nt = 0; nt < NT; ++nt)
                acc[mt][nt] = __builtin_amdgcn_wmma_f32_16x16x32_bf16(
                    false, af[mt].v, false, bfr[nt].v,
                    (short)0, acc[mt][nt], false, false);
    }

    // ---- epilogue: D layout: VGPR i -> row (laneH*8 + i), col laneN ----
#pragma unroll
    for (int mt = 0; mt < MT; ++mt) {
#pragma unroll
        for (int nt = 0; nt < NT; ++nt) {
            int baseR = m0 + waveM * 64 + mt * 16 + laneH * 8;
            int c     = n0 + waveN * 64 + nt * 16 + laneN;
            float bv  = bias ? bias[c] : 0.0f;
#pragma unroll
            for (int i = 0; i < 8; ++i) {
                int r = baseR + i;
                float v = acc[mt][nt][i] * alpha + bv;
                if (doGelu) v = 0.5f * v * (1.0f + erff(v * 0.70710678118f));
                size_t idx = (size_t)r * ldc + c;
                if (resid) v += resid[idx];
                if (Cf) Cf[idx] = v;
                if (Cb) Cb[idx] = (bf16)v;
            }
        }
    }
}

// ---------------------------------------------------------------------------
// LayerNorm (fp32 in, bf16 out) — one block per row of H elements.
// ---------------------------------------------------------------------------
__global__ __launch_bounds__(256)
void ln_bf16_kernel(const float* __restrict__ x,
                    const float* __restrict__ w,
                    const float* __restrict__ b,
                    bf16* __restrict__ out, int H)
{
    const int row = blockIdx.x;
    const int tid = threadIdx.x;
    const float* xr = x + (size_t)row * H;
    bf16* orow = out + (size_t)row * H;

    __shared__ float r0[256];
    __shared__ float r1[256];

    float s = 0.0f, s2 = 0.0f;
    for (int i = tid; i < H; i += 256) {
        float v = xr[i];
        s += v; s2 += v * v;
    }
    r0[tid] = s; r1[tid] = s2;
    __syncthreads();
    for (int o = 128; o > 0; o >>= 1) {
        if (tid < o) { r0[tid] += r0[tid + o]; r1[tid] += r1[tid + o]; }
        __syncthreads();
    }
    float mu  = r0[0] / (float)H;
    float var = r1[0] / (float)H - mu * mu;
    float rs  = rsqrtf(var + 1e-6f);
    for (int i = tid; i < H; i += 256)
        orow[i] = (bf16)((xr[i] - mu) * rs * w[i] + b[i]);
}

// ---------------------------------------------------------------------------
// Row softmax (fp32 scores in, bf16 probs out) — one block per row.
// ---------------------------------------------------------------------------
__global__ __launch_bounds__(256)
void softmax_bf16_kernel(const float* __restrict__ S,
                         bf16* __restrict__ P, int Ncol)
{
    const int row = blockIdx.x;
    const int tid = threadIdx.x;
    const float* sr = S + (size_t)row * Ncol;
    bf16* pr = P + (size_t)row * Ncol;

    __shared__ float red[256];

    float m = -3.4e38f;
    for (int i = tid; i < Ncol; i += 256) m = fmaxf(m, sr[i]);
    red[tid] = m; __syncthreads();
    for (int o = 128; o > 0; o >>= 1) {
        if (tid < o) red[tid] = fmaxf(red[tid], red[tid + o]);
        __syncthreads();
    }
    m = red[0];
    __syncthreads();

    float s = 0.0f;
    for (int i = tid; i < Ncol; i += 256) s += __expf(sr[i] - m);
    red[tid] = s; __syncthreads();
    for (int o = 128; o > 0; o >>= 1) {
        if (tid < o) red[tid] += red[tid + o];
        __syncthreads();
    }
    float inv = 1.0f / red[0];
    for (int i = tid; i < Ncol; i += 256)
        pr[i] = (bf16)(__expf(sr[i] - m) * inv);
}

// ---------------------------------------------------------------------------
// fp32 -> bf16 convert (for weight matrices).
// ---------------------------------------------------------------------------
__global__ __launch_bounds__(256)
void f2bf_kernel(const float* __restrict__ in, bf16* __restrict__ out, size_t n)
{
    size_t i      = (size_t)blockIdx.x * blockDim.x + threadIdx.x;
    size_t stride = (size_t)gridDim.x * blockDim.x;
    for (; i < n; i += stride) out[i] = (bf16)in[i];
}

// ---------------------------------------------------------------------------
// Launch sequence
// ---------------------------------------------------------------------------
extern "C" void kernel_launch(void* const* d_in, const int* in_sizes, int n_in,
                              void* d_out, int out_size, void* d_ws, size_t ws_size,
                              hipStream_t stream)
{
    (void)in_sizes; (void)n_in; (void)out_size; (void)ws_size;

    constexpr int Bb = 4, Nn = 2048, H = 1024, I = 4096;
    constexpr int M = Bb * Nn;               // 8192 rows

    const float* x     = (const float*)d_in[0];
    const float* ln1w  = (const float*)d_in[1];
    const float* ln1b  = (const float*)d_in[2];
    const float* ln2w  = (const float*)d_in[3];
    const float* ln2b  = (const float*)d_in[4];
    const float* qkvw  = (const float*)d_in[5];
    const float* qkvb  = (const float*)d_in[6];
    const float* projw = (const float*)d_in[7];
    const float* projb = (const float*)d_in[8];
    const float* fc1w  = (const float*)d_in[9];
    const float* fc1b  = (const float*)d_in[10];
    const float* fc2w  = (const float*)d_in[11];
    const float* fc2b  = (const float*)d_in[12];

    size_t off = 0;
    auto alloc = [&](size_t bytes) -> void* {
        void* p = (char*)d_ws + off;
        off += (bytes + 255) & ~(size_t)255;
        return p;
    };
    bf16*  wqkv = (bf16*) alloc((size_t)3 * H * H * 2);
    bf16*  wprj = (bf16*) alloc((size_t)H * H * 2);
    bf16*  wfc1 = (bf16*) alloc((size_t)I * H * 2);
    bf16*  wfc2 = (bf16*) alloc((size_t)H * I * 2);
    bf16*  hn   = (bf16*) alloc((size_t)M * H * 2);        // reused for LN1 and LN2
    bf16*  qkv  = (bf16*) alloc((size_t)M * 3 * H * 2);    // [q|k|v] columns
    float* sc   = (float*)alloc((size_t)Bb * Nn * Nn * 4); // attention scores
    bf16*  p    = (bf16*) alloc((size_t)Bb * Nn * Nn * 2); // softmax probs
    bf16*  ctx  = (bf16*) alloc((size_t)M * H * 2);
    float* h    = (float*)alloc((size_t)M * H * 4);        // post-attn residual
    bf16*  ff   = (bf16*) alloc((size_t)M * I * 2);

    // weights -> bf16
    f2bf_kernel<<<4096, 256, 0, stream>>>(qkvw, wqkv, (size_t)3 * H * H);
    f2bf_kernel<<<4096, 256, 0, stream>>>(projw, wprj, (size_t)H * H);
    f2bf_kernel<<<4096, 256, 0, stream>>>(fc1w, wfc1, (size_t)I * H);
    f2bf_kernel<<<4096, 256, 0, stream>>>(fc2w, wfc2, (size_t)H * I);

    // LN1
    ln_bf16_kernel<<<M, 256, 0, stream>>>(x, ln1w, ln1b, hn, H);

    // QKV: [M,3H] = hn[M,H] @ qkv_w[3H,H]^T + qkv_b    (bf16 out only)
    gemm_wmma_kernel<false><<<dim3(3 * H / BN, M / BM, 1), 256, 0, stream>>>(
        hn, H, 0, wqkv, H, 0, qkvb, nullptr,
        nullptr, qkv, 3 * H, 0, H, 1.0f, 0);

    // scores[b] = (q_b @ k_b^T) * H^-0.5   (batched over z; fp32 out)
    gemm_wmma_kernel<false><<<dim3(Nn / BN, Nn / BM, Bb), 256, 0, stream>>>(
        qkv /*q*/, 3 * H, (long long)Nn * 3 * H,
        qkv + H /*k*/, 3 * H, (long long)Nn * 3 * H,
        nullptr, nullptr,
        sc, nullptr, Nn, (long long)Nn * Nn,
        H, 0.03125f /* 1/sqrt(1024) */, 0);

    // softmax rows -> bf16 probs
    softmax_bf16_kernel<<<Bb * Nn, 256, 0, stream>>>(sc, p, Nn);

    // ctx[b] = P_b[2048,2048] @ v_b[2048,1024]   (B not transposed -> TRANS_B)
    gemm_wmma_kernel<true><<<dim3(H / BN, Nn / BM, Bb), 256, 0, stream>>>(
        p, Nn, (long long)Nn * Nn,
        qkv + 2 * H /*v*/, 3 * H, (long long)Nn * 3 * H,
        nullptr, nullptr,
        nullptr, ctx, H, (long long)Nn * H,
        Nn, 1.0f, 0);

    // h = x + ctx @ proj_w^T + proj_b   (fp32 out)
    gemm_wmma_kernel<false><<<dim3(H / BN, M / BM, 1), 256, 0, stream>>>(
        ctx, H, 0, wprj, H, 0, projb, x,
        h, nullptr, H, 0, H, 1.0f, 0);

    // LN2
    ln_bf16_kernel<<<M, 256, 0, stream>>>(h, ln2w, ln2b, hn, H);

    // ff = gelu(hn @ fc1_w^T + fc1_b)   (bf16 out)
    gemm_wmma_kernel<false><<<dim3(I / BN, M / BM, 1), 256, 0, stream>>>(
        hn, H, 0, wfc1, H, 0, fc1b, nullptr,
        nullptr, ff, I, 0, H, 1.0f, 1);

    // out = h + ff @ fc2_w^T + fc2_b    (fp32 final output)
    gemm_wmma_kernel<false><<<dim3(H / BN, M / BM, 1), 256, 0, stream>>>(
        ff, I, 0, wfc2, I, 0, fc2b, h,
        (float*)d_out, nullptr, H, 0, I, 1.0f, 0);
}
